// GCNRegression_13597866459795
// MI455X (gfx1250) — matrix operand order repaired and checked by reading, not compiled, not run on hardware
//
#include <hip/hip_runtime.h>
#include <hip/hip_bf16.h>

typedef __attribute__((ext_vector_type(16))) __bf16 v16bf;
typedef __attribute__((ext_vector_type(8)))  float  v8f;
typedef unsigned short u16;
typedef unsigned int   u32;

union Frag16 { v16bf v; uint4 q[2]; };

__device__ __forceinline__ u16 f32_to_bf16_rne(float f) {
    u32 b = __float_as_uint(f);
    b += 0x7FFFu + ((b >> 16) & 1u);   // round-to-nearest-even
    return (u16)(b >> 16);
}

// ---------------------------------------------------------------- degree / norm
__global__ void k_init_deg(float* deg, int n) {
    int i = blockIdx.x * blockDim.x + threadIdx.x;
    if (i < n) deg[i] = 1.0f;          // self loop
}

__global__ void k_deg_edges(const int* __restrict__ col, float* deg, int nE) {
    int e = blockIdx.x * blockDim.x + threadIdx.x;
    if (e < nE) atomicAdd(&deg[col[e]], 1.0f);
}

__global__ void k_rsqrt_inplace(float* deg, int n) {
    int i = blockIdx.x * blockDim.x + threadIdx.x;
    if (i < n) deg[i] = rsqrtf(deg[i]);
}

// ---------------------------------------------------------------- dtype helpers
__global__ void k_f32_to_bf16(const float* __restrict__ in, u16* __restrict__ out, long n) {
    long i = (long)blockIdx.x * blockDim.x + threadIdx.x;
    if (i < n) out[i] = f32_to_bf16_rne(in[i]);
}

__global__ void k_zero_f32(float* p, long n) {
    long i = (long)blockIdx.x * blockDim.x + threadIdx.x;
    if (i < n) p[i] = 0.0f;
}

// Pre-swizzle W1,W2 (128x128 f32 row-major, W[k][n]) into the CDNA5 16-bit
// B-matrix wave32 layout: flat id = ((((w*8 + tileN)*4 + kstep)*32) + lane)*16 + e
// lane 0-15 -> N=lane, K=kstep*32 + e (e<8), kstep*32 + 16 + (e-8) (e>=8) ; lane 16-31 -> same N, K += 8.
__global__ void k_swizzleW(const float* __restrict__ W1,
                           const float* __restrict__ W2,
                           u16* __restrict__ out) {
    int id = blockIdx.x * blockDim.x + threadIdx.x;   // 32768 total
    int e    =  id        & 15;
    int lane = (id >> 4)  & 31;
    int kk   = (id >> 9)  & 3;
    int t    = (id >> 11) & 7;
    int w    =  id >> 14;
    const float* W = w ? W2 : W1;
    int n = t * 16 + (lane & 15);
    int g = lane >> 4;
    int k = kk * 32 + 8 * g + (e & 7) + ((e >> 3) << 4);
    out[id] = f32_to_bf16_rne(W[k * 128 + n]);
}

// ---------------------------------------------------------------- WMMA GEMM
// C[n,128] = A[n,128](bf16) @ Bsw(128x128 swizzled bf16), f32 accumulate.
// Each wave owns one 16-row strip x all 128 cols (8 accumulator tiles).
__global__ __launch_bounds__(256) void k_gemm_bf16(
    const u16* __restrict__ A, const u16* __restrict__ Bsw,
    float* __restrict__ C, int nNodes) {
    const int lane = threadIdx.x & 31;
    const int tileM = (blockIdx.x * 8 + (threadIdx.x >> 5)) * 16;
    if (tileM >= nNodes) return;                  // wave-uniform: EXEC stays full
    const int g = lane >> 4;
    int row = tileM + (lane & 15);
    if (row >= nNodes) row = nNodes - 1;          // clamp; masked on store
    const u16* arow = A + (long)row * 128;

    v8f acc[8] = {};
    #pragma unroll
    for (int kk = 0; kk < 4; ++kk) {
        Frag16 a;
        const int kb = kk * 32 + 8 * g;           // A layout: runs kb..kb+7, kb+16..kb+23
        a.q[0] = *(const uint4*)(arow + kb);
        a.q[1] = *(const uint4*)(arow + kb + 16);
        #pragma unroll
        for (int t = 0; t < 8; ++t) {
            Frag16 b;
            const u16* bp = Bsw + (((t * 4 + kk) * 32 + lane) * 16);
            b.q[0] = *(const uint4*)(bp);
            b.q[1] = *(const uint4*)(bp + 8);
            acc[t] = __builtin_amdgcn_wmma_f32_16x16x32_bf16(
                false, a.v, false, b.v, (short)0, acc[t], false, false);
        }
    }
    #pragma unroll
    for (int t = 0; t < 8; ++t) {
        const int n = t * 16 + (lane & 15);
        #pragma unroll
        for (int r = 0; r < 8; ++r) {
            const int m = tileM + r + 8 * g;
            if (m < nNodes) C[(long)m * 128 + n] = acc[t][r];
        }
    }
}

// ---------------------------------------------------------------- edge scatter
// One wave per edge: 32 lanes x float4 = 128 channels. Feature rows are L2
// resident (51.2MB << 192MB L2), so gather + atomic scatter is L2-bound.
__global__ __launch_bounds__(256) void k_edge_agg(
    const int* __restrict__ row, const int* __restrict__ col,
    const float* __restrict__ dis, const float* __restrict__ xw,
    float* __restrict__ agg, int nE) {
    const int wave = (int)((blockIdx.x * (long)blockDim.x + threadIdx.x) >> 5);
    const int lane = threadIdx.x & 31;
    if (wave >= nE) return;
    const int r = row[wave], c = col[wave];
    // hint next edge's source row toward the caches (global_prefetch_b8)
    if (lane == 0 && wave + 8 < nE)
        __builtin_prefetch(xw + (long)row[wave + 8] * 128, 0, 1);
    const float nrm = dis[r] * dis[c];
    const float4 v = *(const float4*)(xw + (long)r * 128 + lane * 4);
    float* dst = agg + (long)c * 128 + lane * 4;
    atomicAdd(dst + 0, v.x * nrm);
    atomicAdd(dst + 1, v.y * nrm);
    atomicAdd(dst + 2, v.z * nrm);
    atomicAdd(dst + 3, v.w * nrm);
}

// ---------------------------------------------------------------- finalize L1
// h = relu(agg + xw*dis^2 + b); emit bf16 for next GEMM's A operand.
__global__ void k_finalize_relu_bf16(
    const float* __restrict__ agg, const float* __restrict__ xw,
    const float* __restrict__ dis, const float* __restrict__ bias,
    u16* __restrict__ hbf, int nNodes) {
    long i = (long)blockIdx.x * blockDim.x + threadIdx.x;
    if (i >= (long)nNodes * 128) return;
    const int node = (int)(i >> 7), c = (int)(i & 127);
    const float d = dis[node];
    float h = agg[i] + xw[i] * d * d + bias[c];
    hbf[i] = f32_to_bf16_rne(fmaxf(h, 0.0f));
}

// ---------------------------------------------------------------- finalize L2 + readout
// One wave per node: h2 = relu(agg + xw*dis^2 + b2), out = dot(h2, Wout)+bout.
__global__ __launch_bounds__(256) void k_finalize_out(
    const float* __restrict__ agg, const float* __restrict__ xw,
    const float* __restrict__ dis, const float* __restrict__ b2,
    const float* __restrict__ Wout, const float* __restrict__ bout,
    float* __restrict__ out, int nNodes) {
    const int node = (int)((blockIdx.x * (long)blockDim.x + threadIdx.x) >> 5);
    const int lane = threadIdx.x & 31;
    if (node >= nNodes) return;
    const float d = dis[node], dd = d * d;
    const long base = (long)node * 128 + lane * 4;
    const float4 a = *(const float4*)(agg + base);
    const float4 x = *(const float4*)(xw + base);
    const int c = lane * 4;
    float s;
    s  = fmaxf(a.x + x.x * dd + b2[c + 0], 0.0f) * Wout[c + 0];
    s += fmaxf(a.y + x.y * dd + b2[c + 1], 0.0f) * Wout[c + 1];
    s += fmaxf(a.z + x.z * dd + b2[c + 2], 0.0f) * Wout[c + 2];
    s += fmaxf(a.w + x.w * dd + b2[c + 3], 0.0f) * Wout[c + 3];
    #pragma unroll
    for (int off = 16; off > 0; off >>= 1) s += __shfl_down(s, off, 32);
    if (lane == 0) out[node] = s + bout[0];
}

// ================================================================ launch
extern "C" void kernel_launch(void* const* d_in, const int* in_sizes, int n_in,
                              void* d_out, int out_size, void* d_ws, size_t ws_size,
                              hipStream_t stream) {
    const float* x    = (const float*)d_in[0];
    const int*   ei   = (const int*)  d_in[1];
    const float* W1   = (const float*)d_in[2];
    const float* b1   = (const float*)d_in[3];
    const float* W2   = (const float*)d_in[4];
    const float* b2   = (const float*)d_in[5];
    const float* Wout = (const float*)d_in[6];
    const float* bout = (const float*)d_in[7];
    float* out = (float*)d_out;

    const int nNodes = in_sizes[0] / 128;
    const int nEdges = in_sizes[1] / 2;
    const int* rows = ei;
    const int* cols = ei + nEdges;
    const long feat = (long)nNodes * 128;

    // workspace carve-up (~128.5 MB for N=100K)
    char* p = (char*)d_ws;
    float* bufA = (float*)p;  p += feat * sizeof(float);   // GEMM outputs
    float* bufB = (float*)p;  p += feat * sizeof(float);   // aggregation
    u16*   hbf  = (u16*)p;    p += feat * sizeof(u16);     // bf16 activations
    u16*   wsw  = (u16*)p;    p += 2L * 128 * 128 * sizeof(u16); // swizzled W1,W2
    float* dis  = (float*)p;  p += (long)nNodes * sizeof(float);
    (void)ws_size; (void)n_in; (void)out_size;

    const int T = 256;
    const int gNode  = (nNodes + T - 1) / T;
    const int gEdge  = (nEdges + T - 1) / T;
    const int gFeat  = (int)((feat + T - 1) / T);
    const int gGemm  = ((nNodes + 15) / 16 + 7) / 8;      // 8 waves/block
    const int gEdgeW = (int)(((long)nEdges * 32 + T - 1) / T);
    const int gNodeW = (int)(((long)nNodes * 32 + T - 1) / T);

    // ---- degrees -> dis = rsqrt(deg)
    k_init_deg     <<<gNode, T, 0, stream>>>(dis, nNodes);
    k_deg_edges    <<<gEdge, T, 0, stream>>>(cols, dis, nEdges);
    k_rsqrt_inplace<<<gNode, T, 0, stream>>>(dis, nNodes);

    // ---- precision / layout prep
    k_f32_to_bf16<<<gFeat, T, 0, stream>>>(x, hbf, feat);
    k_swizzleW   <<<128,   T, 0, stream>>>(W1, W2, wsw);

    // ---- layer 1
    k_gemm_bf16<<<gGemm, T, 0, stream>>>(hbf, wsw, bufA, nNodes);
    k_zero_f32 <<<gFeat, T, 0, stream>>>(bufB, feat);
    k_edge_agg <<<gEdgeW, T, 0, stream>>>(rows, cols, dis, bufA, bufB, nEdges);
    k_finalize_relu_bf16<<<gFeat, T, 0, stream>>>(bufB, bufA, dis, b1, hbf, nNodes);

    // ---- layer 2
    k_gemm_bf16<<<gGemm, T, 0, stream>>>(hbf, wsw + 8 * 4 * 32 * 16, bufA, nNodes);
    k_zero_f32 <<<gFeat, T, 0, stream>>>(bufB, feat);
    k_edge_agg <<<gEdgeW, T, 0, stream>>>(rows, cols, dis, bufA, bufB, nEdges);

    // ---- fused finalize + readout
    k_finalize_out<<<gNodeW, T, 0, stream>>>(bufB, bufA, dis, b2, Wout, bout, out, nNodes);
}
